// DPGLTransformer_27513560498639
// MI455X (gfx1250) — matrix operand order repaired and checked by reading, not compiled, not run on hardware
//
#include <hip/hip_runtime.h>
#include <hip/hip_bf16.h>
#include <stdint.h>

// ---------------- CDNA5 (gfx1250) wave32 WMMA types ----------------
typedef __bf16 v16bf __attribute__((ext_vector_type(16)));
typedef __bf16 v8bf  __attribute__((ext_vector_type(8)));
typedef float  v8f   __attribute__((ext_vector_type(8)));

#define WSZ     7
#define N_TOK   49
#define HEADS   16
#define DIM     512
#define HDIM    32
#define BWIN    4096
#define NWIN    1024
#define MROWS   (BWIN * N_TOK)     // 200704 rows, divisible by 32

// ---- Tensor Data Mover availability (probe-verified builtin) ----
#if defined(__has_builtin)
#  if __has_builtin(__builtin_amdgcn_tensor_load_to_lds)
#    define HAVE_TDM 1
#  endif
#endif
#ifdef HAVE_TDM
typedef unsigned int u32x4 __attribute__((ext_vector_type(4)));
typedef int          i32x8 __attribute__((ext_vector_type(8)));
typedef int          i32x4 __attribute__((ext_vector_type(4)));
// therock-10 headers ship only with the 6-arg (clang-23) toolchain
#  if __has_include(<hip/amd_detail/amd_gfx1250_TDM.h>)
#    define TDM_SIX_ARGS 1
#  endif
#endif

__device__ __forceinline__ __bf16 f2bf(float f) {
    unsigned u = __builtin_bit_cast(unsigned, f);
    unsigned r = u + 0x7FFFu + ((u >> 16) & 1u);      // round-to-nearest-even
    return __builtin_bit_cast(__bf16, (unsigned short)(r >> 16));
}

__device__ __forceinline__ v8bf ld8(const __bf16* p) {
    return __builtin_bit_cast(v8bf, *(const uint4*)p);   // b128 load (16B aligned)
}
__device__ __forceinline__ v16bf cat8(v8bf lo, v8bf hi) {
    return __builtin_shufflevector(lo, hi, 0,1,2,3,4,5,6,7,8,9,10,11,12,13,14,15);
}

// ===================================================================
// One-time f32 -> bf16 weight conversion (weights reused 6272x by the
// GEMMs; convert once so the WMMA hot loop is pure load+wmma).
// ===================================================================
__global__ __launch_bounds__(256)
void cvt_bf16(const float* __restrict__ src, __bf16* __restrict__ dst)
{
    const int i = (blockIdx.x * 256 + threadIdx.x) * 8;
    const float4 a = *(const float4*)(src + i);
    const float4 b = *(const float4*)(src + i + 4);
    v8bf v;
    v[0] = f2bf(a.x); v[1] = f2bf(a.y); v[2] = f2bf(a.z); v[3] = f2bf(a.w);
    v[4] = f2bf(b.x); v[5] = f2bf(b.y); v[6] = f2bf(b.z); v[7] = f2bf(b.w);
    *(uint4*)(dst + i) = __builtin_bit_cast(uint4, v);
}

// ===================================================================
// 512x512 GEMM: Y[m,o] = sum_k X[m,k] * Wb[o,k] + bias[o]  (torch Linear)
// Block: 32 rows x 512 cols, K=512. 8 waves; wave -> (rowTile, 8 nTiles).
// A staged in LDS (bf16); all 16 A K-fragments hoisted into VGPRs.
// ===================================================================
template<bool IN_BF16, bool OUT_QKV>
__global__ __launch_bounds__(256)
void gemm512(const void* __restrict__ xin, const __bf16* __restrict__ Wb,
             const float* __restrict__ bias, void* __restrict__ yout, float scale)
{
    constexpr int LDA = 520;                      // halves; 1040B row = 16B-multiple
    __shared__ __bf16 smA[32 * LDA];

    const int m0   = blockIdx.x * 32;
    const int t    = threadIdx.x;
    const int lane = t & 31;
    const int wave = t >> 5;
    const int lmod = lane & 15;
    const int lhi  = lane >> 4;                   // lane-half select

    // ---- stage A tile (32 x 512) into LDS as bf16 ----
    {
        const int r  = t >> 3;                    // 0..31
        const int c0 = (t & 7) * 64;              // 8 threads per row
        if (IN_BF16) {
            const __bf16* x = (const __bf16*)xin + (size_t)(m0 + r) * DIM + c0;
            #pragma unroll
            for (int c = 0; c < 64; c += 8)
                *(uint4*)(&smA[r * LDA + c0 + c]) = *(const uint4*)(x + c);
        } else {
            const float* x = (const float*)xin + (size_t)(m0 + r) * DIM + c0;
            #pragma unroll
            for (int c = 0; c < 64; c += 4) {
                float4 v = *(const float4*)(x + c);
                __bf16* d = &smA[r * LDA + c0 + c];
                d[0] = f2bf(v.x); d[1] = f2bf(v.y); d[2] = f2bf(v.z); d[3] = f2bf(v.w);
            }
        }
    }
    __syncthreads();

    const int rt  = wave & 1;                     // row tile 0/1 (16 rows each)
    const int nt0 = (wave >> 1) * 8;              // 8 output tiles per wave

    // ---- hoist all 16 A K-fragments for this wave's row tile ----
    v16bf afr[16];
    #pragma unroll
    for (int ks = 0; ks < 16; ++ks) {
        const __bf16* ap = &smA[(rt * 16 + lmod) * LDA + ks * 32 + lhi * 16];
        afr[ks] = cat8(ld8(ap), ld8(ap + 8));
    }

    for (int nti = 0; nti < 8; ++nti) {
        const int nt   = nt0 + nti;
        const int ocol = nt * 16 + lmod;
        const __bf16* wrow = Wb + (size_t)ocol * DIM + lhi * 16;
        __builtin_prefetch(wrow + DIM * 16, 0, 1);          // next tile's rows
        v8f acc = {};
        #pragma unroll
        for (int ks = 0; ks < 16; ++ks) {
            const __bf16* wp = wrow + ks * 32;
            v16bf b = cat8(ld8(wp), ld8(wp + 8));
            acc = __builtin_amdgcn_wmma_f32_16x16x32_bf16(false, afr[ks], false, b,
                                                          (short)0, acc, false, false);
        }
        const float bo = bias[ocol];
        if (OUT_QKV) {  // -> bf16 [b, h, n, d] with (y+bias)*scale
            int m   = m0 + rt * 16 + lhi * 8;
            int bw  = m / N_TOK;
            int nt2 = m - bw * N_TOK;
            const int h = ocol >> 5, dd = ocol & 31;
            #pragma unroll
            for (int r = 0; r < 8; ++r) {
                ((__bf16*)yout)[(((size_t)bw * HEADS + h) * N_TOK + nt2) * HDIM + dd] =
                    f2bf((acc[r] + bo) * scale);
                if (++nt2 == N_TOK) { nt2 = 0; ++bw; }       // incremental /49
            }
        } else {        // -> f32 [m, o]
            #pragma unroll
            for (int r = 0; r < 8; ++r) {
                const int m = m0 + rt * 16 + lhi * 8 + r;
                ((float*)yout)[(size_t)m * DIM + ocol] = acc[r] + bo;
            }
        }
    }
}

// ===================================================================
// Windowed attention: one wave per (window b, head h). 49 padded to 64.
// S = Q K^T (16 WMMAs) + rel-pos bias + mask (staged in LDS; the mask
// window is DMA'd by the Tensor Data Mover when available). Softmax is
// reduction-free: unnormalized e^s goes to LDS, and the row sums are
// produced by an extra WMMA against an all-ones B fragment (every lane
// of Pexp x Ones holds its own rows' sums) -- no ds_bpermute at all.
// Normalization is folded into the output scaling.
// ===================================================================
__global__ __launch_bounds__(128)
void attn49(const __bf16* __restrict__ Q, const __bf16* __restrict__ K,
            const __bf16* __restrict__ V, const float* __restrict__ bias_table,
            const float* __restrict__ mask, __bf16* __restrict__ X)
{
    constexpr int LDP = 64;                       // 128B rows -> 16B-aligned b128
    constexpr int LDV = 64;
    __shared__ __bf16 smP [4][64 * LDP];          // 32 KB
    __shared__ __bf16 smVt[4][32 * LDV];          // 16 KB
    __shared__ float  smMask[N_TOK * N_TOK + 3];  // 9.6 KB (per-block window)
    __shared__ float  smBias[4][176];             // 2.8 KB (per-wave head column)

    const int b    = blockIdx.x >> 2;
    const int hg   = blockIdx.x & 3;
    const int wave = threadIdx.x >> 5;
    const int lane = threadIdx.x & 31;
    const int h    = hg * 4 + wave;
    const int lmod = lane & 15;
    const int lhi  = lane >> 4;

    const size_t bh = ((size_t)b * HEADS + h) * N_TOK;
    const __bf16* Qh = Q + bh * HDIM;
    const __bf16* Kh = K + bh * HDIM;
    const __bf16* Vh = V + bh * HDIM;
    __bf16* P  = smP [wave];
    __bf16* Vt = smVt[wave];
    float*  biasw = smBias[wave];
    const float* maskw = mask + (size_t)(b & (NWIN - 1)) * N_TOK * N_TOK;

    // ---- mask window -> LDS: TDM DMA when available, else cooperative ----
#ifdef HAVE_TDM
    if (wave == 0) {
        const unsigned long long ga = (unsigned long long)(uintptr_t)maskw;
        const unsigned lds = (unsigned)(size_t)(const void*)smMask;
        u32x4 g0 = { 1u,                                   // count=1 (valid user D#)
                     lds,                                  // lds_addr
                     (unsigned)ga,                         // global_addr[31:0]
                     (unsigned)((ga >> 32) & 0x01FFFFFFu)  // global_addr[56:32]
                       | 0x80000000u };                    // type=2 ("image")
        i32x8 g1 = { (int)0x20000,          // data_size=2 (4B)
                     (int)(2401u << 16),    // tensor_dim0[15:0]=2401
                     (int)(1u  << 16),      // dim0 hi=0, tensor_dim1=1
                     (int)(2401u << 16),    // dim1 hi=0, tile_dim0=2401
                     1,                     // tile_dim1=1
                     2401, 0, 0 };          // tensor_dim0_stride=2401
        i32x4 z4 = { 0, 0, 0, 0 };
#ifdef TDM_SIX_ARGS
        i32x8 z8 = { 0, 0, 0, 0, 0, 0, 0, 0 };
        __builtin_amdgcn_tensor_load_to_lds(g0, g1, z4, z4, z8, 0);
#else
        __builtin_amdgcn_tensor_load_to_lds(g0, g1, z4, z4, 0);
#endif
    }
#if __has_builtin(__builtin_amdgcn_s_wait_tensorcnt)
    __builtin_amdgcn_s_wait_tensorcnt(0);
#else
    asm volatile("s_wait_tensorcnt 0x0" ::: "memory");
#endif
#else
    for (int ii = threadIdx.x; ii < N_TOK * N_TOK; ii += 128) smMask[ii] = maskw[ii];
#endif

    // ---- per-head relative-position-bias column -> LDS (wave-private) ----
    for (int idx = lane; idx < 169; idx += 32) biasw[idx] = bias_table[idx * HEADS + h];
    __syncthreads();

    // ---- transpose V (49x32 -> Vt[d][m], zero-padded to m=64) ----
    const __bf16 zb = __builtin_bit_cast(__bf16, (unsigned short)0);
    for (int i = lane; i < 32 * LDV; i += 32) Vt[i] = zb;
    for (int m = lane; m < N_TOK; m += 32) {
        const __bf16* vr = Vh + (size_t)m * HDIM;
        #pragma unroll
        for (int c = 0; c < 4; ++c) {
            v8bf v = ld8(vr + c * 8);
            #pragma unroll
            for (int d = 0; d < 8; ++d) Vt[(c * 8 + d) * LDV + m] = v[d];
        }
    }

    // ---- hoisted per-lane key-column math (depends only on j, lane) ----
    int  mcl[4], cj[4];
    bool mok[4];
    #pragma unroll
    for (int j = 0; j < 4; ++j) {
        const int m = j * 16 + lmod;
        mok[j] = (m < N_TOK);
        mcl[j] = mok[j] ? m : N_TOK - 1;
        const int ym = mcl[j] / WSZ;
        const int xm = mcl[j] - ym * WSZ;
        // idx = (yn*13 + xn) + cj ; cj = 84 - ym*13 - xm
        cj[j] = (WSZ - 1) * (2 * WSZ - 1) + (WSZ - 1) - ym * (2 * WSZ - 1) - xm;
    }

    // ---- hoisted K fragments (independent of query band i) ----
    v16bf kb[4];
    #pragma unroll
    for (int j = 0; j < 4; ++j) {
        const __bf16* kp = Kh + (size_t)mcl[j] * HDIM + lhi * 16;
        kb[j] = cat8(ld8(kp), ld8(kp + 8));
    }

    // ---- scores -> unnormalized e^s (no reductions, exp is f32-safe) ----
    for (int i = 0; i < 4; ++i) {
        const int nrow = i * 16 + lmod;
        const int nq   = nrow < N_TOK ? nrow : N_TOK - 1;   // clamp pad rows
        const __bf16* qp = Qh + (size_t)nq * HDIM + lhi * 16;
        const v16bf qa = cat8(ld8(qp), ld8(qp + 8));

        v8f S[4];
        #pragma unroll
        for (int j = 0; j < 4; ++j) {
            v8f z = {};
            S[j] = __builtin_amdgcn_wmma_f32_16x16x32_bf16(false, qa, false, kb[j],
                                                           (short)0, z, false, false);
        }
        // one division per band; (rb = yn*13+xn, xn, row base) advance incrementally
        int ncc = i * 16 + lhi * 8;
        if (ncc > N_TOK - 1) ncc = N_TOK - 1;
        const int yn0 = ncc / WSZ;
        int xn = ncc - yn0 * WSZ;
        int rb = yn0 * (2 * WSZ - 1) + xn;
        const float* mrow = smMask + ncc * N_TOK;
        #pragma unroll
        for (int r = 0; r < 8; ++r) {
            #pragma unroll
            for (int j = 0; j < 4; ++j) {
                const float add = biasw[rb + cj[j]] + mrow[mcl[j]];
                const float e   = mok[j] ? __expf(S[j][r] + add) : 0.0f;
                P[(i * 16 + lhi * 8 + r) * LDP + j * 16 + lmod] = f2bf(e);
            }
            // advance clamped row coordinates
            if (ncc < N_TOK - 1) {
                ++ncc; mrow += N_TOK;
                const bool w = (xn == WSZ - 1);
                rb += w ? WSZ : 1;                 // +13-6 on wrap, else +1
                xn  = w ? 0 : xn + 1;
            }
        }
    }

    // ---- hoisted V fragments + all-ones row-sum fragment ----
    v16bf vb[2][2];
    #pragma unroll
    for (int dt = 0; dt < 2; ++dt)
        #pragma unroll
        for (int ks = 0; ks < 2; ++ks) {
            const __bf16* vp = &Vt[(dt * 16 + lmod) * LDV + ks * 32 + lhi * 16];
            vb[dt][ks] = cat8(ld8(vp), ld8(vp + 8));
        }
    const __bf16 one_b = __builtin_bit_cast(__bf16, (unsigned short)0x3F80);
    v16bf ones;
    #pragma unroll
    for (int e = 0; e < 16; ++e) ones[e] = one_b;

    // ---- out = (Pexp x V) * (1 / (Pexp x Ones)) ; rows n<49 -> bf16 X ----
    for (int i = 0; i < 4; ++i) {
        v8f acc0 = {}, acc1 = {}, accS = {};
        #pragma unroll
        for (int ks = 0; ks < 2; ++ks) {
            const __bf16* pp = &P[(i * 16 + lmod) * LDP + ks * 32 + lhi * 16];
            const v16bf pa = cat8(ld8(pp), ld8(pp + 8));
            acc0 = __builtin_amdgcn_wmma_f32_16x16x32_bf16(false, pa, false, vb[0][ks],
                                                           (short)0, acc0, false, false);
            acc1 = __builtin_amdgcn_wmma_f32_16x16x32_bf16(false, pa, false, vb[1][ks],
                                                           (short)0, acc1, false, false);
            accS = __builtin_amdgcn_wmma_f32_16x16x32_bf16(false, pa, false, ones,
                                                           (short)0, accS, false, false);
        }
        #pragma unroll
        for (int r = 0; r < 8; ++r) {
            const int n = i * 16 + lhi * 8 + r;
            if (n < N_TOK) {
                const float inv = 1.0f / accS[r];      // row sum (> 0 for valid rows)
                __bf16* xr = X + ((size_t)b * N_TOK + n) * DIM + h * HDIM + lmod;
                xr[0]  = f2bf(acc0[r] * inv);
                xr[16] = f2bf(acc1[r] * inv);
            }
        }
    }
}

// ===================================================================
extern "C" void kernel_launch(void* const* d_in, const int* in_sizes, int n_in,
                              void* d_out, int out_size, void* d_ws, size_t ws_size,
                              hipStream_t stream)
{
    const float* local_f    = (const float*)d_in[0];
    const float* global_f   = (const float*)d_in[1];
    const float* mask       = (const float*)d_in[2];
    const float* bias_table = (const float*)d_in[3];
    const float* Wk = (const float*)d_in[4];
    const float* bk = (const float*)d_in[5];
    const float* Wq = (const float*)d_in[6];
    const float* bq = (const float*)d_in[7];
    const float* Wv = (const float*)d_in[8];
    const float* bv = (const float*)d_in[9];
    const float* Wp = (const float*)d_in[10];
    const float* bp = (const float*)d_in[11];

    const size_t QKV_ELEMS = (size_t)BWIN * HEADS * N_TOK * HDIM;  // 102,760,448
    const size_t W_ELEMS   = (size_t)DIM * DIM;                    // 262,144
    __bf16* dQ  = (__bf16*)d_ws;
    __bf16* dK  = dQ + QKV_ELEMS;
    __bf16* dV  = dK + QKV_ELEMS;
    __bf16* dX  = dV + QKV_ELEMS;
    __bf16* dWk = dX + QKV_ELEMS;
    __bf16* dWq = dWk + W_ELEMS;
    __bf16* dWv = dWq + W_ELEMS;
    __bf16* dWp = dWv + W_ELEMS;

    // one-time weight conversion: 262144 / (256 threads * 8 elems) = 128 blocks
    cvt_bf16<<<128, 256, 0, stream>>>(Wk, dWk);
    cvt_bf16<<<128, 256, 0, stream>>>(Wq, dWq);
    cvt_bf16<<<128, 256, 0, stream>>>(Wv, dWv);
    cvt_bf16<<<128, 256, 0, stream>>>(Wp, dWp);

    const float qscale = 0.17677669529663687f;    // 32^-0.5
    const dim3 gG(MROWS / 32);                    // 6272 blocks

    gemm512<false, true ><<<gG, 256, 0, stream>>>(local_f,  dWk, bk, dK, 1.0f);
    gemm512<false, true ><<<gG, 256, 0, stream>>>(global_f, dWq, bq, dQ, qscale);
    gemm512<false, true ><<<gG, 256, 0, stream>>>(local_f,  dWv, bv, dV, 1.0f);
    attn49<<<BWIN * 4, 128, 0, stream>>>(dQ, dK, dV, bias_table, mask, dX);
    gemm512<true,  false><<<gG, 256, 0, stream>>>(dX, dWp, bp, (float*)d_out, 1.0f);

    (void)in_sizes; (void)n_in; (void)out_size; (void)ws_size;
}
